// CausalSelfAttention_26603027431528
// MI455X (gfx1250) — matrix operand order repaired and checked by reading, not compiled, not run on hardware
//
#include <hip/hip_runtime.h>

typedef _Float16 half_t;
typedef __attribute__((ext_vector_type(16))) _Float16 v16h;
typedef __attribute__((ext_vector_type(8)))  _Float16 v8h;
typedef __attribute__((ext_vector_type(8)))  float    v8f;

union V16H { v16h v; v8h h[2]; };

// ---------------------------------------------------------------------------
// WMMA fragment helpers (CDNA5 wave32, V_WMMA_F32_16X16X32_F16)
// A 16x32 (f16): lane l: m = l&15, g = l>>4; halves 0-7  = K g*8+0..7,
//                halves 8-15 = K 16+g*8+0..7   -> two contiguous 16B loads.
// B 32x16 (f16): lane l: n = l&15, kh = l>>4;  halves 0-7 = K kh*8+0..7,
//                halves 8-15 = K 16+kh*8+0..7 -> two 16B loads from row n.
// C/D 16x16 f32: lane l: n = l&15, row = r + 8*(l>>4), r = vgpr 0..7.
// ---------------------------------------------------------------------------
__device__ inline v16h load_a_frag(const half_t* __restrict__ base, int ld, int lane) {
    int m = lane & 15, g = lane >> 4;
    const half_t* p = base + (long)m * ld + g * 8;
    V16H r;
    r.h[0] = *(const v8h*)(p);
    r.h[1] = *(const v8h*)(p + 16);
    return r.v;
}

__device__ inline v16h load_b_frag(const half_t* __restrict__ base, int ld, int lane) {
    int n = lane & 15, kh = lane >> 4;
    const half_t* p = base + (long)n * ld + kh * 8;
    V16H r;
    r.h[0] = *(const v8h*)(p);
    r.h[1] = *(const v8h*)(p + 16);
    return r.v;
}

__device__ inline v8f wmma_f16(v16h a, v16h b, v8f c) {
    return __builtin_amdgcn_wmma_f32_16x16x32_f16(false, a, false, b, (short)0, c, false, false);
}

// ---------------------------------------------------------------------------
// f32 -> f16 elementwise convert
// ---------------------------------------------------------------------------
__global__ void conv_f32_f16(const float* __restrict__ src, half_t* __restrict__ dst, int n) {
    int i = blockIdx.x * blockDim.x + threadIdx.x;
    int stride = gridDim.x * blockDim.x;
    for (; i < n; i += stride) dst[i] = (half_t)src[i];
}

// ---------------------------------------------------------------------------
// GEMM: C[m,n] = sum_d A[m,d] * W[n,d]   (A: [8192,2048] f16, W: [N,2048] f16)
// block = 128 threads (4 waves); block computes 128x64 tile; wave w owns rows
// [m0, m0+32) x 64 cols = 2 A frags x 4 B frags -> 8 WMMAs per 12 b128 loads.
// Working set < 192MB L2, so direct global loads (no LDS staging) are fine.
// mode 0: Q -> Qh[b][h=16][s][128] f16      mode 1: K -> Kh[b][kv=4][s][128] f16
// mode 2: V -> Vt[b][kv=4][128][s] f16      mode 3: O -> out f32 row-major
// ---------------------------------------------------------------------------
__global__ void gemm_wmma(const half_t* __restrict__ A, const half_t* __restrict__ W,
                          half_t* __restrict__ outH, float* __restrict__ outF, int mode) {
    const int lane = threadIdx.x & 31;
    const int wave = threadIdx.x >> 5;
    const int m0 = blockIdx.x * 128 + wave * 32;
    const int n0 = blockIdx.y * 64;

    v8f acc[2][4] = {};
    for (int k = 0; k < 2048; k += 32) {
        v16h a0 = load_a_frag(A + (long)m0 * 2048 + k, 2048, lane);
        v16h a1 = load_a_frag(A + (long)(m0 + 16) * 2048 + k, 2048, lane);
#pragma unroll
        for (int t = 0; t < 4; ++t) {
            v16h b = load_b_frag(W + (long)(n0 + t * 16) * 2048 + k, 2048, lane);
            acc[0][t] = wmma_f16(a0, b, acc[0][t]);
            acc[1][t] = wmma_f16(a1, b, acc[1][t]);
        }
    }

    const int nl = lane & 15, hf = lane >> 4;
#pragma unroll
    for (int u = 0; u < 2; ++u) {
#pragma unroll
        for (int t = 0; t < 4; ++t) {
#pragma unroll
            for (int r = 0; r < 8; ++r) {
                int m = m0 + u * 16 + 8 * hf + r;   // global row (b*S + s)
                int e = n0 + t * 16 + nl;           // global output column
                float v = acc[u][t][r];
                int b = m >> 11, s = m & 2047;
                if (mode == 0) {                    // Q: [b][h][s][d]
                    int h = e >> 7, d = e & 127;
                    outH[(((long)(b * 16 + h) * 2048 + s) << 7) + d] = (half_t)v;
                } else if (mode == 1) {             // K: [b][kv][s][d]
                    int h = e >> 7, d = e & 127;
                    outH[(((long)(b * 4 + h) * 2048 + s) << 7) + d] = (half_t)v;
                } else if (mode == 2) {             // V transposed: [b][kv][d][s]
                    int h = e >> 7, d = e & 127;
                    outH[(((long)(b * 4 + h) * 128 + d) << 11) + s] = (half_t)v;
                } else {                            // O: f32 [b*S][2048]
                    outF[(long)m * 2048 + e] = v;
                }
            }
        }
    }
}

// ---------------------------------------------------------------------------
// In-place RoPE on [nheads][2048][128] f16 (head-major). One thread per
// (head, s, d<64) pair; reads both halves, writes both -> in-place safe.
// ---------------------------------------------------------------------------
__global__ void rope_kernel(half_t* __restrict__ p) {
    int i = blockIdx.x * blockDim.x + threadIdx.x;
    int d = i & 63;
    int s = (i >> 6) & 2047;
    int hl = i >> 17;                           // 2048*64 = 2^17 per head
    half_t* base = p + (((long)hl * 2048 + s) << 7);
    // inv_freq = 10000^(-2d/128) = exp(-(2d/128)*ln(10000))
    float inv = __expf(-(float)(2 * d) * (9.210340371976184f / 128.0f));
    float ang = (float)s * inv;
    float sn, cs;
    __sincosf(ang, &sn, &cs);
    float x1 = (float)base[d], x2 = (float)base[d + 64];
    base[d]      = (half_t)(x1 * cs + x2 * sn);
    base[d + 64] = (half_t)(-x1 * sn + x2 * cs);
}

// ---------------------------------------------------------------------------
// Flash attention, causal, GQA rep=4. Block = 4 waves; each wave owns one
// 16-query tile, iterates keys in steps of 32 (two 16x16 logit tiles/step).
// Online softmax in f32; P re-shaped f32 D-frag -> f16 A-frag via LDS.
// ---------------------------------------------------------------------------
__global__ void attn_kernel(const half_t* __restrict__ Qh, const half_t* __restrict__ Kh,
                            const half_t* __restrict__ Vt, half_t* __restrict__ Y) {
    __shared__ __align__(64) half_t Pbuf[4][16 * 32];

    const int lane = threadIdx.x & 31;
    const int wave = threadIdx.x >> 5;
    const int h = blockIdx.y, b = blockIdx.z;
    const int qt = blockIdx.x * 4 + wave;
    const int q0 = qt * 16;
    const int kv = h >> 2;

    const half_t* Qbase = Qh + (((long)(b * 16 + h) * 2048) << 7);
    const half_t* Kbase = Kh + (((long)(b * 4 + kv) * 2048) << 7);
    const half_t* Vbase = Vt + (((long)(b * 4 + kv) * 128) << 11);

    const int nl = lane & 15, hf = lane >> 4;

    // Q fragments for d = 0,32,64,96 with 1/sqrt(128) folded in
    v16h qf[4];
    const _Float16 sc = (_Float16)0.08838834764831845f;
#pragma unroll
    for (int t = 0; t < 4; ++t) {
        V16H r;
        int m = nl, g = hf;
        const half_t* p = Qbase + ((long)(q0 + m) << 7) + t * 32 + g * 8;
        r.h[0] = *(const v8h*)(p);
        r.h[1] = *(const v8h*)(p + 16);
#pragma unroll
        for (int i = 0; i < 16; ++i) r.v[i] = r.v[i] * sc;
        qf[t] = r.v;
    }

    v8f acc[8] = {};
    float mrow[8], lrow[8];
#pragma unroll
    for (int r = 0; r < 8; ++r) { mrow[r] = -1e30f; lrow[r] = 0.f; }

    const int qlast = q0 + 15;
    half_t* pb = &Pbuf[wave][0];

    for (int k0 = 0; k0 <= qlast; k0 += 32) {
        // logits: two 16x16 tiles (keys k0..k0+15 and k0+16..k0+31)
        v8f s0 = {}, s1 = {};
#pragma unroll
        for (int t = 0; t < 4; ++t) {
            v16h b0 = load_b_frag(Kbase + ((long)k0 << 7) + t * 32, 128, lane);
            s0 = wmma_f16(qf[t], b0, s0);
            v16h b1 = load_b_frag(Kbase + ((long)(k0 + 16) << 7) + t * 32, 128, lane);
            s1 = wmma_f16(qf[t], b1, s1);
        }

        // causal mask + per-row max candidate
        float pm[8];
#pragma unroll
        for (int r = 0; r < 8; ++r) {
            int q = q0 + 8 * hf + r;
            float a0 = (k0 + nl <= q) ? s0[r] : -1e30f;
            float a1 = (k0 + 16 + nl <= q) ? s1[r] : -1e30f;
            s0[r] = a0; s1[r] = a1;
            pm[r] = fmaxf(a0, a1);
        }
        // reduce max across the 16 lanes holding one row
#pragma unroll
        for (int off = 1; off < 16; off <<= 1) {
#pragma unroll
            for (int r = 0; r < 8; ++r) pm[r] = fmaxf(pm[r], __shfl_xor(pm[r], off));
        }

        float alpha[8], lsum[8];
#pragma unroll
        for (int r = 0; r < 8; ++r) {
            float mn = fmaxf(mrow[r], pm[r]);
            alpha[r] = __expf(mrow[r] - mn);
            mrow[r] = mn;
            float p0 = __expf(s0[r] - mn);
            float p1 = __expf(s1[r] - mn);
            s0[r] = p0; s1[r] = p1;
            lsum[r] = p0 + p1;
#pragma unroll
            for (int t = 0; t < 8; ++t) acc[t][r] *= alpha[r];
        }
#pragma unroll
        for (int off = 1; off < 16; off <<= 1) {
#pragma unroll
            for (int r = 0; r < 8; ++r) lsum[r] += __shfl_xor(lsum[r], off);
        }
#pragma unroll
        for (int r = 0; r < 8; ++r) lrow[r] = lrow[r] * alpha[r] + lsum[r];

        // P (f32 D-frag) -> LDS -> f16 A-frag
#pragma unroll
        for (int r = 0; r < 8; ++r) {
            int m = 8 * hf + r;
            pb[m * 32 + nl]      = (half_t)s0[r];
            pb[m * 32 + 16 + nl] = (half_t)s1[r];
        }
        asm volatile("s_wait_dscnt 0" ::: "memory");
        v16h pa = load_a_frag(pb, 32, lane);

        // acc += P * V  (8 d-tiles of 16)
#pragma unroll
        for (int t = 0; t < 8; ++t) {
            v16h vb = load_b_frag(Vbase + ((long)(t * 16) << 11) + k0, 2048, lane);
            acc[t] = wmma_f16(pa, vb, acc[t]);
        }
    }

    // normalize and write Y[b][q][h*128 + d] (f16)
#pragma unroll
    for (int r = 0; r < 8; ++r) lrow[r] = 1.0f / lrow[r];
#pragma unroll
    for (int t = 0; t < 8; ++t) {
#pragma unroll
        for (int r = 0; r < 8; ++r) {
            int q = q0 + 8 * hf + r;
            int d = t * 16 + nl;
            Y[(((long)b * 2048 + q) << 11) + h * 128 + d] = (half_t)(acc[t][r] * lrow[r]);
        }
    }
}

// ---------------------------------------------------------------------------
// Host-side orchestration
// ---------------------------------------------------------------------------
extern "C" void kernel_launch(void* const* d_in, const int* in_sizes, int n_in,
                              void* d_out, int out_size, void* d_ws, size_t ws_size,
                              hipStream_t stream) {
    const float* x  = (const float*)d_in[0];
    const float* wq = (const float*)d_in[1];
    const float* wk = (const float*)d_in[2];
    const float* wv = (const float*)d_in[3];
    const float* wo = (const float*)d_in[4];
    float* out = (float*)d_out;

    char* ws = (char*)d_ws;
    half_t* Xh  = (half_t*)(ws);                       // 32 MiB  [8192][2048]
    half_t* Wqh = (half_t*)(ws + (size_t)33554432);    //  8 MiB  [2048][2048]
    half_t* Wkh = (half_t*)(ws + (size_t)41943040);    //  2 MiB  [512][2048]
    half_t* Wvh = (half_t*)(ws + (size_t)44040192);    //  2 MiB  [512][2048]
    half_t* Woh = (half_t*)(ws + (size_t)46137344);    //  8 MiB  [2048][2048]
    half_t* Qh  = (half_t*)(ws + (size_t)54525952);    // 32 MiB  [4][16][2048][128]
    half_t* Kh  = (half_t*)(ws + (size_t)88080384);    //  8 MiB  [4][4][2048][128]
    half_t* Vt  = (half_t*)(ws + (size_t)96468992);    //  8 MiB  [4][4][128][2048]
    half_t* Y   = (half_t*)(ws + (size_t)104857600);   // 32 MiB  [8192][2048]

    // f32 -> f16 converts
    conv_f32_f16<<<8192, 256, 0, stream>>>(x,  Xh,  16777216);
    conv_f32_f16<<<8192, 256, 0, stream>>>(wq, Wqh, 4194304);
    conv_f32_f16<<<4096, 256, 0, stream>>>(wk, Wkh, 1048576);
    conv_f32_f16<<<4096, 256, 0, stream>>>(wv, Wvh, 1048576);
    conv_f32_f16<<<8192, 256, 0, stream>>>(wo, Woh, 4194304);

    // QKV projections (WMMA): grid.x = 8192/128 row-blocks
    gemm_wmma<<<dim3(64, 32), 128, 0, stream>>>(Xh, Wqh, Qh, nullptr, 0);
    gemm_wmma<<<dim3(64, 8),  128, 0, stream>>>(Xh, Wkh, Kh, nullptr, 1);
    gemm_wmma<<<dim3(64, 8),  128, 0, stream>>>(Xh, Wvh, Vt, nullptr, 2);

    // RoPE in place: Q has 64 head-planes, K has 16
    rope_kernel<<<32768, 256, 0, stream>>>(Qh);   // 64*2048*64 threads
    rope_kernel<<<8192,  256, 0, stream>>>(Kh);   // 16*2048*64 threads

    // causal flash attention (WMMA)
    attn_kernel<<<dim3(32, 16, 4), 128, 0, stream>>>(Qh, Kh, Vt, Y);

    // output projection -> f32 out (WMMA)
    gemm_wmma<<<dim3(64, 32), 128, 0, stream>>>(Y, Woh, nullptr, out, 3);
}
